// AdditiveAttention_15607911154069
// MI455X (gfx1250) — matrix operand order repaired and checked by reading, not compiled, not run on hardware
//
#include <hip/hip_runtime.h>
#include <math.h>

#define BB 4
#define SS 4096
#define DD 512

typedef __attribute__((ext_vector_type(16))) __bf16 v16bf;
typedef __attribute__((ext_vector_type(8)))  __bf16 v8bf;
typedef __attribute__((ext_vector_type(8)))  float  v8f;

#define WMMA_BF16(a, b, c) \
  (c) = __builtin_amdgcn_wmma_f32_16x16x32_bf16(false, (a), false, (b), (short)0, (c), false, false)

// A-fragment (16x32, MxK) from an f32 row-major matrix, converting to bf16 in-register.
// row already includes (lane&15).
__device__ __forceinline__ v16bf load_a_f32(const float* __restrict__ src, long row,
                                            int lda, int k0, int lane) {
  const float* p = src + row * (long)lda + k0 + ((lane >> 4) << 3);
  float4 c0 = *(const float4*)(p);
  float4 c1 = *(const float4*)(p + 4);
  float4 c2 = *(const float4*)(p + 16);
  float4 c3 = *(const float4*)(p + 20);
  v16bf a;
  a[0]  = (__bf16)c0.x; a[1]  = (__bf16)c0.y; a[2]  = (__bf16)c0.z; a[3]  = (__bf16)c0.w;
  a[4]  = (__bf16)c1.x; a[5]  = (__bf16)c1.y; a[6]  = (__bf16)c1.z; a[7]  = (__bf16)c1.w;
  a[8]  = (__bf16)c2.x; a[9]  = (__bf16)c2.y; a[10] = (__bf16)c2.z; a[11] = (__bf16)c2.w;
  a[12] = (__bf16)c3.x; a[13] = (__bf16)c3.y; a[14] = (__bf16)c3.z; a[15] = (__bf16)c3.w;
  return a;
}

// A-fragment from a bf16 row-major matrix.
__device__ __forceinline__ v16bf load_a_bf16(const __bf16* __restrict__ src, long row,
                                             int lda, int k0, int lane) {
  const __bf16* p = src + row * (long)lda + k0 + ((lane >> 4) << 3);
  v8bf c0 = *(const v8bf*)(p);
  v8bf c1 = *(const v8bf*)(p + 16);
  v16bf a;
#pragma unroll
  for (int j = 0; j < 8; ++j) { a[j] = c0[j]; a[8 + j] = c1[j]; }
  return a;
}

// B-fragment (32x16, KxN) from a bf16 [n][k] matrix (K contiguous per n).
// n already includes (lane&15).
__device__ __forceinline__ v16bf load_b_bf16(const __bf16* __restrict__ src, int n,
                                             int ldb, int k0, int lane) {
  const __bf16* p = src + (long)n * ldb + k0 + ((lane >> 4) << 3);
  v8bf c0 = *(const v8bf*)(p);
  v8bf c1 = *(const v8bf*)(p + 16);
  v16bf b;
#pragma unroll
  for (int j = 0; j < 8; ++j) { b[j] = c0[j]; b[8 + j] = c1[j]; }
  return b;
}

// Software-pipelined K-pass: acc[0..3] += A(16xK) * B(Kx64), A from f32 source.
// n0 = first of 4 consecutive 16-wide N tiles. Double-buffers A and B fragments
// so iteration i+1's loads overlap iteration i's WMMAs.
__device__ __forceinline__ void gemm_pass_f32A(const float* __restrict__ A, long arow, int lda,
                                               const __bf16* __restrict__ Bm, int n0, int ldb,
                                               int K, int lane, v8f* acc) {
  int ncol = lane & 15;
  v16bf a_c = load_a_f32(A, arow, lda, 0, lane);
  v16bf b_c[4];
#pragma unroll
  for (int j = 0; j < 4; ++j) b_c[j] = load_b_bf16(Bm, n0 + j * 16 + ncol, ldb, 0, lane);
#pragma unroll 2
  for (int k0 = 32; k0 < K; k0 += 32) {
    v16bf a_n = load_a_f32(A, arow, lda, k0, lane);
    v16bf b_n[4];
#pragma unroll
    for (int j = 0; j < 4; ++j) b_n[j] = load_b_bf16(Bm, n0 + j * 16 + ncol, ldb, k0, lane);
#pragma unroll
    for (int j = 0; j < 4; ++j) WMMA_BF16(a_c, b_c[j], acc[j]);
    a_c = a_n;
#pragma unroll
    for (int j = 0; j < 4; ++j) b_c[j] = b_n[j];
  }
#pragma unroll
  for (int j = 0; j < 4; ++j) WMMA_BF16(a_c, b_c[j], acc[j]);
}

// Same, A from bf16 source.
__device__ __forceinline__ void gemm_pass_bf16A(const __bf16* __restrict__ A, long arow, int lda,
                                                const __bf16* __restrict__ Bm, int n0, int ldb,
                                                int K, int lane, v8f* acc) {
  int ncol = lane & 15;
  v16bf a_c = load_a_bf16(A, arow, lda, 0, lane);
  v16bf b_c[4];
#pragma unroll
  for (int j = 0; j < 4; ++j) b_c[j] = load_b_bf16(Bm, n0 + j * 16 + ncol, ldb, 0, lane);
#pragma unroll 2
  for (int k0 = 32; k0 < K; k0 += 32) {
    v16bf a_n = load_a_bf16(A, arow, lda, k0, lane);
    v16bf b_n[4];
#pragma unroll
    for (int j = 0; j < 4; ++j) b_n[j] = load_b_bf16(Bm, n0 + j * 16 + ncol, ldb, k0, lane);
#pragma unroll
    for (int j = 0; j < 4; ++j) WMMA_BF16(a_c, b_c[j], acc[j]);
    a_c = a_n;
#pragma unroll
    for (int j = 0; j < 4; ++j) b_c[j] = b_n[j];
  }
#pragma unroll
  for (int j = 0; j < 4; ++j) WMMA_BF16(a_c, b_c[j], acc[j]);
}

// ---------------- K0: f32 -> bf16 weight conversion ----------------
__global__ void cvt_f32_bf16(const float* __restrict__ src, __bf16* __restrict__ dst, int n) {
  int i = blockIdx.x * blockDim.x + threadIdx.x;
  if (i < n) dst[i] = (__bf16)src[i];
}

// ---------------- K1: V = value @ Wv^T, stored transposed [b][e][t] ----------------
__global__ void vgemm_kernel(const float* __restrict__ value, const __bf16* __restrict__ wvb,
                             __bf16* __restrict__ vtb) {
  int wave = threadIdx.x >> 5, lane = threadIdx.x & 31;
  long r0 = (long)blockIdx.x * 16;
  int b  = (int)(r0 / SS);
  int t0 = (int)(r0 % SS);
  long arow = r0 + (lane & 15);
  int ncol = lane & 15;
  int mb = (lane >> 4) * 8;
  v8f acc[4] = {};
  gemm_pass_f32A(value, arow, DD, wvb, wave * 64, DD, DD, lane, acc);
#pragma unroll
  for (int j = 0; j < 4; ++j) {
    int e = wave * 64 + j * 16 + ncol;
    v8bf pk;
#pragma unroll
    for (int i = 0; i < 8; ++i) pk[i] = (__bf16)acc[j][i];
    *(v8bf*)(vtb + ((long)b * DD + e) * SS + t0 + mb) = pk;  // transposed, 16B contiguous
  }
}

// ---------------- K2: scores = tanh(q·Wq^T + k·Wk^T), mask -> attn buffer ----------------
__global__ void scores_kernel(const float* __restrict__ query, const float* __restrict__ key,
                              const unsigned char* __restrict__ mask,
                              const __bf16* __restrict__ wqb, const __bf16* __restrict__ wkb,
                              float* __restrict__ attn) {
  int wave = threadIdx.x >> 5, lane = threadIdx.x & 31;
  long r0 = (long)blockIdx.x * 16;
  long arow = r0 + (lane & 15);
  int ncol = lane & 15;
  int mb = (lane >> 4) * 8;
  for (int g = wave; g < SS / 64; g += 8) {   // each wave: groups of 4 N-tiles
    v8f acc[4] = {};
    // Two pipelined passes into the same accumulators: scores = tanh(Q + K)
    gemm_pass_f32A(query, arow, DD, wqb, g * 64, DD, DD, lane, acc);
    gemm_pass_f32A(key,   arow, DD, wkb, g * 64, DD, DD, lane, acc);
#pragma unroll
    for (int j = 0; j < 4; ++j) {
      int t = g * 64 + j * 16 + ncol;
#pragma unroll
      for (int i = 0; i < 8; ++i) {
        long idx = (r0 + mb + i) * SS + t;
        float v = tanhf(acc[j][i]);
        if (mask[idx]) v = -1e10f;
        attn[idx] = v;
      }
    }
  }
}

// ---------------- K3: row softmax over last dim (in place) ----------------
__global__ void softmax_kernel(float* __restrict__ attn) {
  __shared__ float red[256];
  float* p = attn + (long)blockIdx.x * SS;
  int tid = threadIdx.x;
  float vals[16];
  float m = -3.4e38f;
#pragma unroll
  for (int i = 0; i < 16; ++i) { vals[i] = p[tid + i * 256]; m = fmaxf(m, vals[i]); }
  red[tid] = m; __syncthreads();
  for (int off = 128; off > 0; off >>= 1) {
    if (tid < off) red[tid] = fmaxf(red[tid], red[tid + off]);
    __syncthreads();
  }
  m = red[0]; __syncthreads();
  float s = 0.f;
#pragma unroll
  for (int i = 0; i < 16; ++i) { vals[i] = __expf(vals[i] - m); s += vals[i]; }
  red[tid] = s; __syncthreads();
  for (int off = 128; off > 0; off >>= 1) {
    if (tid < off) red[tid] += red[tid + off];
    __syncthreads();
  }
  float inv = 1.0f / red[0];
#pragma unroll
  for (int i = 0; i < 16; ++i) p[tid + i * 256] = vals[i] * inv;
}

// ---------------- K4: context = attn @ V (A from f32 attn, B from V^T bf16) ----------------
__global__ void ctx_kernel(const float* __restrict__ attn, const __bf16* __restrict__ vtb,
                           __bf16* __restrict__ ctxb) {
  int wave = threadIdx.x >> 5, lane = threadIdx.x & 31;
  long r0 = (long)blockIdx.x * 16;
  int b = (int)(r0 / SS);
  const __bf16* vt = vtb + (long)b * DD * SS;
  long arow = r0 + (lane & 15);
  int ncol = lane & 15;
  int mb = (lane >> 4) * 8;
  v8f acc[4] = {};
  gemm_pass_f32A(attn, arow, SS, vt, wave * 64, SS, SS, lane, acc);
#pragma unroll
  for (int j = 0; j < 4; ++j) {
    int e = wave * 64 + j * 16 + ncol;
#pragma unroll
    for (int i = 0; i < 8; ++i) ctxb[(r0 + mb + i) * DD + e] = (__bf16)acc[j][i];
  }
}

// ---------------- K5: out = context @ Wfc^T + residual(value) ----------------
__global__ void out_kernel(const __bf16* __restrict__ ctxb, const __bf16* __restrict__ wfcb,
                           const float* __restrict__ value, float* __restrict__ out) {
  int wave = threadIdx.x >> 5, lane = threadIdx.x & 31;
  long r0 = (long)blockIdx.x * 16;
  long arow = r0 + (lane & 15);
  int ncol = lane & 15;
  int mb = (lane >> 4) * 8;
  v8f acc[4] = {};
  gemm_pass_bf16A(ctxb, arow, DD, wfcb, wave * 64, DD, DD, lane, acc);
#pragma unroll
  for (int j = 0; j < 4; ++j) {
    int e = wave * 64 + j * 16 + ncol;
#pragma unroll
    for (int i = 0; i < 8; ++i) {
      long idx = (r0 + mb + i) * DD + e;
      out[idx] = acc[j][i] + value[idx];
    }
  }
}

extern "C" void kernel_launch(void* const* d_in, const int* in_sizes, int n_in,
                              void* d_out, int out_size, void* d_ws, size_t ws_size,
                              hipStream_t stream) {
  const float* query = (const float*)d_in[0];
  const float* key_  = (const float*)d_in[1];
  const float* value = (const float*)d_in[2];
  const unsigned char* mask = (const unsigned char*)d_in[3];
  const float* Wq  = (const float*)d_in[4];
  const float* Wk  = (const float*)d_in[5];
  const float* Wv  = (const float*)d_in[6];
  const float* Wfc = (const float*)d_in[7];

  float* out  = (float*)d_out;
  float* attn = out + (long)BB * SS * DD;   // tuple order: (out, attn)

  // workspace layout (bf16): Wq | Wk | Wv | Wfc | V^T[b][e][t] | context[r][e]  (~43 MB)
  __bf16* wqb  = (__bf16*)d_ws;
  __bf16* wkb  = wqb  + (long)SS * DD;
  __bf16* wvb  = wkb  + (long)SS * DD;
  __bf16* wfcb = wvb  + (long)DD * DD;
  __bf16* vtb  = wfcb + (long)DD * DD;
  __bf16* ctxb = vtb  + (long)BB * DD * SS;

  int nw = SS * DD, nv = DD * DD;
  cvt_f32_bf16<<<(nw + 255) / 256, 256, 0, stream>>>(Wq,  wqb,  nw);
  cvt_f32_bf16<<<(nw + 255) / 256, 256, 0, stream>>>(Wk,  wkb,  nw);
  cvt_f32_bf16<<<(nv + 255) / 256, 256, 0, stream>>>(Wv,  wvb,  nv);
  cvt_f32_bf16<<<(nv + 255) / 256, 256, 0, stream>>>(Wfc, wfcb, nv);

  int rb = BB * SS / 16;  // 1024 row-blocks of 16 rows, 256 threads = 8 wave32 each
  vgemm_kernel <<<rb, 256, 0, stream>>>(value, wvb, vtb);
  scores_kernel<<<rb, 256, 0, stream>>>(query, key_, mask, wqb, wkb, attn);
  softmax_kernel<<<BB * SS, 256, 0, stream>>>(attn);
  ctx_kernel   <<<rb, 256, 0, stream>>>(attn, vtb, ctxb);
  out_kernel   <<<rb, 256, 0, stream>>>(ctxb, wfcb, value, out);
}
